// TextEmbeddingQBNNPipeline_20907900797064
// MI455X (gfx1250) — compile-verified, matrix operand order checked by reading, not run on hardware
//
#include <hip/hip_runtime.h>
#include <hip/hip_bf16.h>
#include <math.h>

// ---------------------------------------------------------------------------
// TextEmbeddingQBNNPipeline for MI455X (gfx1250, wave32, WMMA).
// All large GEMMs use v_wmma_f32_16x16x32_f16 (f16 inputs, f32 accum).
// ---------------------------------------------------------------------------

typedef __attribute__((ext_vector_type(16))) _Float16 v16h;
typedef __attribute__((ext_vector_type(8)))  _Float16 v8h;
typedef __attribute__((ext_vector_type(8)))  float    v8f;

#define LL 4
#define DD 1024
#define HH 16
#define HD 64
#define FF 4096
#define VV 32000
#define SS 1024

// ---------------------------------------------------------------------------
// Block reduction helper (256 threads)
// ---------------------------------------------------------------------------
__device__ __forceinline__ float blk_sum(float v, float* red) {
    int tid = threadIdx.x;
    red[tid] = v;
    __syncthreads();
    #pragma unroll
    for (int off = 128; off > 0; off >>= 1) {
        if (tid < off) red[tid] += red[tid + off];
        __syncthreads();
    }
    float r = red[0];
    __syncthreads();
    return r;
}

// ---------------------------------------------------------------------------
// Embedding gather + LayerNorm (one block per token, 256 threads)
// ---------------------------------------------------------------------------
__global__ __launch_bounds__(256)
void k_embed_ln(const float* __restrict__ tok, const float* __restrict__ pos,
                const float* __restrict__ g, const float* __restrict__ bta,
                const int* __restrict__ ids, float* __restrict__ X) {
    int t = blockIdx.x;            // 0..NT-1
    int s = t % SS;
    int id = ids[t];
    __shared__ float tile[DD];
    __shared__ float red[256];
    float sum = 0.f;
    for (int d = threadIdx.x; d < DD; d += 256) {
        float v = tok[(size_t)id * DD + d] + pos[(size_t)s * DD + d];
        tile[d] = v; sum += v;
    }
    float mean = blk_sum(sum, red) * (1.0f / DD);
    float sq = 0.f;
    for (int d = threadIdx.x; d < DD; d += 256) {
        float c = tile[d] - mean; sq += c * c;
    }
    float var = blk_sum(sq, red) * (1.0f / DD);
    float rs = rsqrtf(var + 1e-5f);
    for (int d = threadIdx.x; d < DD; d += 256)
        X[(size_t)t * DD + d] = (tile[d] - mean) * rs * g[d] + bta[d];
}

// ---------------------------------------------------------------------------
// Generic LayerNorm over width W (W <= 4096), one block per token
// ---------------------------------------------------------------------------
__global__ __launch_bounds__(256)
void k_ln(const float* __restrict__ In, const float* __restrict__ g,
          const float* __restrict__ bta, float* __restrict__ Out, int W) {
    int t = blockIdx.x;
    __shared__ float tile[FF];
    __shared__ float red[256];
    float sum = 0.f;
    for (int d = threadIdx.x; d < W; d += 256) {
        float v = In[(size_t)t * W + d];
        tile[d] = v; sum += v;
    }
    float mean = blk_sum(sum, red) / (float)W;
    float sq = 0.f;
    for (int d = threadIdx.x; d < W; d += 256) {
        float c = tile[d] - mean; sq += c * c;
    }
    float var = blk_sum(sq, red) / (float)W;
    float rs = rsqrtf(var + 1e-5f);
    for (int d = threadIdx.x; d < W; d += 256)
        Out[(size_t)t * W + d] = (tile[d] - mean) * rs * g[d] + bta[d];
}

// ---------------------------------------------------------------------------
// QBNN post: y = gelu( LN( ht + lam * sJ * tanh(ht) ) ), width W (<=4096)
// ---------------------------------------------------------------------------
__global__ __launch_bounds__(256)
void k_qbnn_post(const float* __restrict__ Ht, const float* __restrict__ SJ,
                 const float* __restrict__ lam_arr, int layer,
                 const float* __restrict__ g, const float* __restrict__ bb,
                 float* __restrict__ Out, int W) {
    int t = blockIdx.x;
    float lam = lam_arr[layer];
    __shared__ float tile[FF];
    __shared__ float red[256];
    float sum = 0.f;
    for (int d = threadIdx.x; d < W; d += 256) {
        float h = Ht[(size_t)t * W + d];
        float v = h + lam * SJ[(size_t)t * W + d] * tanhf(h);
        tile[d] = v; sum += v;
    }
    float mean = blk_sum(sum, red) / (float)W;
    float sq = 0.f;
    for (int d = threadIdx.x; d < W; d += 256) {
        float c = tile[d] - mean; sq += c * c;
    }
    float var = blk_sum(sq, red) / (float)W;
    float rs = rsqrtf(var + 1e-5f);
    for (int d = threadIdx.x; d < W; d += 256) {
        float y = (tile[d] - mean) * rs * g[d] + bb[d];
        Out[(size_t)t * W + d] = 0.5f * y * (1.0f + erff(y * 0.70710678118654752f));
    }
}

// ---------------------------------------------------------------------------
// Elementwise residual add: X += F
// ---------------------------------------------------------------------------
__global__ void k_add(float* __restrict__ X, const float* __restrict__ F, int n) {
    int i = blockIdx.x * blockDim.x + threadIdx.x;
    if (i < n) X[i] += F[i];
}

// ---------------------------------------------------------------------------
// Per-head qj = tanh(q) @ J[h]   (J: HD x HD)
// grid = (NT/16, H), block = 256. Each thread: 1 row x 4 cols.
// ---------------------------------------------------------------------------
__global__ __launch_bounds__(256)
void k_qj(const float* __restrict__ Q, const float* __restrict__ Jl,
          float* __restrict__ QJ) {
    int h = blockIdx.y;
    const float* Jh = Jl + (size_t)h * HD * HD;
    __shared__ float Js[HD * HD];
    for (int idx = threadIdx.x; idx < HD * HD; idx += 256) Js[idx] = Jh[idx];
    __syncthreads();
    int r = threadIdx.x >> 4;                  // 0..15
    int c = (threadIdx.x & 15) * 4;            // 0..60 step 4
    int t = blockIdx.x * 16 + r;
    const float* qrow = Q + (size_t)t * DD + h * HD;
    float a0 = 0.f, a1 = 0.f, a2 = 0.f, a3 = 0.f;
    #pragma unroll 4
    for (int dd = 0; dd < HD; ++dd) {
        float tq = tanhf(qrow[dd]);
        const float* jr = Js + dd * HD + c;
        a0 += tq * jr[0]; a1 += tq * jr[1]; a2 += tq * jr[2]; a3 += tq * jr[3];
    }
    float* o = QJ + (size_t)t * DD + h * HD + c;
    o[0] = a0; o[1] = a1; o[2] = a2; o[3] = a3;
}

// ---------------------------------------------------------------------------
// Streaming causal attention with QBNN delta term (flash-style).
// grid = (B*H, S/32), block = 256. 32 q-rows per block, 64-wide k tiles.
// scores = q.k/8 + lam * qj . tanh(k) ; online softmax; o = P@V / l
// ---------------------------------------------------------------------------
__global__ __launch_bounds__(256)
void k_attn(const float* __restrict__ Q, const float* __restrict__ Kt,
            const float* __restrict__ Vt, const float* __restrict__ QJ,
            const float* __restrict__ lam_arr, int layer,
            float* __restrict__ O) {
    int bh = blockIdx.x;
    int b = bh / HH, h = bh % HH;
    int q0 = blockIdx.y * 32;
    const float scale = 0.125f;                // 1/sqrt(64)
    float lam = lam_arr[layer];

    __shared__ float qL[32][HD];
    __shared__ float qjL[32][HD];
    __shared__ float kv[64][HD];               // k tile, then reused for v tile
    __shared__ float tkL[64][HD];
    __shared__ float sc[32][64];
    __shared__ float mrow[32], lrow[32], crow[32];

    int tid = threadIdx.x;
    for (int idx = tid; idx < 32 * HD; idx += 256) {
        int r = idx >> 6, c = idx & 63;
        size_t t = (size_t)(b * SS + q0 + r) * DD + h * HD + c;
        qL[r][c] = Q[t];
        qjL[r][c] = QJ[t];
    }
    if (tid < 32) { mrow[tid] = -3e38f; lrow[tid] = 0.f; }
    float acc[8];
    #pragma unroll
    for (int u = 0; u < 8; ++u) acc[u] = 0.f;
    int myi = tid >> 3;            // 0..31 (row)
    int myc = (tid & 7) * 8;       // col group
    __syncthreads();

    for (int j0 = 0; j0 <= q0 + 31; j0 += 64) {
        // Load K tile + tanh(K)
        for (int idx = tid; idx < 64 * HD; idx += 256) {
            int r = idx >> 6, c = idx & 63;
            float kvv = Kt[(size_t)(b * SS + j0 + r) * DD + h * HD + c];
            kv[r][c] = kvv;
            tkL[r][c] = tanhf(kvv);
        }
        __syncthreads();
        // Scores
        {
            int si = tid >> 3;
            #pragma unroll
            for (int u = 0; u < 8; ++u) {
                int jj = (tid & 7) * 8 + u;
                int gi = q0 + si, gj = j0 + jj;
                float s;
                if (gj > gi) {
                    s = -3e38f;
                } else {
                    float d1 = 0.f, d2 = 0.f;
                    #pragma unroll 8
                    for (int dd = 0; dd < HD; ++dd) {
                        d1 += qL[si][dd] * kv[jj][dd];
                        d2 += qjL[si][dd] * tkL[jj][dd];
                    }
                    s = scale * d1 + lam * d2;
                }
                sc[si][jj] = s;
            }
        }
        __syncthreads();
        // Reload V tile into kv (k no longer needed)
        for (int idx = tid; idx < 64 * HD; idx += 256) {
            int r = idx >> 6, c = idx & 63;
            kv[r][c] = Vt[(size_t)(b * SS + j0 + r) * DD + h * HD + c];
        }
        // Online softmax row update (one thread per row)
        if (tid < 32) {
            float mx = mrow[tid];
            for (int jj = 0; jj < 64; ++jj) mx = fmaxf(mx, sc[tid][jj]);
            float corr = expf(mrow[tid] - mx);
            float s = 0.f;
            for (int jj = 0; jj < 64; ++jj) {
                float p = expf(sc[tid][jj] - mx);
                sc[tid][jj] = p; s += p;
            }
            lrow[tid] = lrow[tid] * corr + s;
            crow[tid] = corr;
            mrow[tid] = mx;
        }
        __syncthreads();
        // Accumulate P @ V
        {
            float corr = crow[myi];
            #pragma unroll
            for (int u = 0; u < 8; ++u) {
                float sum = 0.f;
                #pragma unroll 8
                for (int jj = 0; jj < 64; ++jj) sum += sc[myi][jj] * kv[jj][myc + u];
                acc[u] = acc[u] * corr + sum;
            }
        }
        __syncthreads();
    }
    float inv = 1.0f / lrow[myi];
    size_t t = (size_t)(b * SS + q0 + myi) * DD + h * HD + myc;
    #pragma unroll
    for (int u = 0; u < 8; ++u) O[t + u] = acc[u] * inv;
}

// ---------------------------------------------------------------------------
// WMMA GEMM: C(MxN) = op(A)(MxK) @ B(KxN) [+bias] [+res]
//   - fp32 in memory, converted to f16 while staging into LDS
//   - tanhA: apply tanh to A elements (implements tanh(h) @ J)
//   - btrans: B stored (NxK) row-major, i.e. use B^T (LM head: x @ lm_W.T)
// Tiles: 128x128x32. 8 waves; each wave 32x64 via 2x4 v_wmma_f32_16x16x32_f16.
// Both tiles staged K-contiguous per row (B stored [N][K]): every fragment
// load is two 16B ds_load_b128 per lane, and every staging write is a 16B
// ds_store_b128 of a packed v8h run (A: lane=M, B: lane=N per ISA 7.12.2;
// C/D: VGPR r -> rows r / r+8 for lane halves).
// Software-pipelined: global loads for tile k+1 (converted to f16 in regs)
// overlap WMMA of tile k; double-buffered LDS, two barriers per k-step.
// ---------------------------------------------------------------------------
#define TM 128
#define TN 128
#define TKK 32
#define KP  40          // padded row (80 B: 16B-aligned, bank-spread)

__global__ __launch_bounds__(256)
void k_gemm(const float* __restrict__ A, const float* __restrict__ B,
            const float* __restrict__ bias, const float* __restrict__ res,
            float* __restrict__ C, int M, int N, int Kd, int tanhA, int btrans) {
    __shared__ __align__(16) _Float16 As[2][TM][KP];
    __shared__ __align__(16) _Float16 Bs[2][TN][KP];

    int n0 = blockIdx.x * TN;
    int m0 = blockIdx.y * TM;
    int tid = threadIdx.x;
    int lane = tid & 31, wave = tid >> 5;
    int waveM = (wave >> 1) * 32;
    int waveN = (wave & 1) * 64;
    int l15 = lane & 15;
    int koff = (lane >> 4) * 8;

    v8f acc[2][4];
    #pragma unroll
    for (int i = 0; i < 2; ++i)
        #pragma unroll
        for (int j = 0; j < 4; ++j)
            acc[i][j] = (v8f){0.f, 0.f, 0.f, 0.f, 0.f, 0.f, 0.f, 0.f};

    // Per-thread staging: two 8-element k-contiguous runs each for A and B,
    // already converted to f16 (v8h) so a run flushes as one ds_store_b128.
    v8h a8[2], b8[2];

    // Run coordinates (A and btrans-B: 512 runs = 128 rows x 4 groups)
    int au0 = tid * 2 + 0, au1 = tid * 2 + 1;
    int aRow0 = au0 >> 2, aCg0 = (au0 & 3) * 8;
    int aRow1 = au1 >> 2, aCg1 = (au1 & 3) * 8;
    // Non-transposed B runs: n fastest across threads for coalescing
    int bn0 = tid & 127,        bKg0 = (tid >> 7) * 8;
    int bn1 = (tid + 256) & 127, bKg1 = ((tid + 256) >> 7) * 8;

    auto gload = [&](int k0) {
        #pragma unroll
        for (int j = 0; j < 2; ++j) {
            int row = j ? aRow1 : aRow0, cg = j ? aCg1 : aCg0;
            const float* p = &A[(size_t)(m0 + row) * Kd + k0 + cg];
            #pragma unroll
            for (int e = 0; e < 8; ++e) {
                float v = p[e];
                a8[j][e] = (_Float16)(tanhA ? tanhf(v) : v);
            }
        }
        if (btrans) {
            #pragma unroll
            for (int j = 0; j < 2; ++j) {
                int row = j ? aRow1 : aRow0, cg = j ? aCg1 : aCg0;
                const float* p = &B[(size_t)(n0 + row) * Kd + k0 + cg];
                #pragma unroll
                for (int e = 0; e < 8; ++e) b8[j][e] = (_Float16)p[e];
            }
        } else {
            #pragma unroll
            for (int j = 0; j < 2; ++j) {
                int n = j ? bn1 : bn0, kg = j ? bKg1 : bKg0;
                #pragma unroll
                for (int e = 0; e < 8; ++e)
                    b8[j][e] = (_Float16)B[(size_t)(k0 + kg + e) * N + n0 + n];
            }
        }
    };
    auto lstore = [&](int buf) {
        *(v8h*)&As[buf][aRow0][aCg0] = a8[0];
        *(v8h*)&As[buf][aRow1][aCg1] = a8[1];
        if (btrans) {
            *(v8h*)&Bs[buf][aRow0][aCg0] = b8[0];
            *(v8h*)&Bs[buf][aRow1][aCg1] = b8[1];
        } else {
            *(v8h*)&Bs[buf][bn0][bKg0] = b8[0];
            *(v8h*)&Bs[buf][bn1][bKg1] = b8[1];
        }
    };

    int nk = Kd / TKK;
    gload(0);
    for (int kt = 0; kt < nk; ++kt) {
        int cur = kt & 1;
        lstore(cur);
        __syncthreads();
        if (kt + 1 < nk) gload((kt + 1) * TKK);           // overlaps WMMA below
        if (kt + 2 < nk) {                                // global_prefetch_b8
            __builtin_prefetch(&A[(size_t)(m0 + aRow0) * Kd + (kt + 2) * TKK + aCg0], 0, 1);
            __builtin_prefetch(btrans
                ? &B[(size_t)(n0 + aRow0) * Kd + (kt + 2) * TKK + aCg0]
                : &B[(size_t)((kt + 2) * TKK + bKg0) * N + n0 + bn0], 0, 1);
        }

        v16h af[2], bf[4];
        #pragma unroll
        for (int ti = 0; ti < 2; ++ti) {
            int mr = waveM + ti * 16 + l15;
            #pragma unroll
            for (int e = 0; e < 16; ++e) {
                int kk = (e < 8 ? e : e + 8) + koff;      // two 16B runs
                af[ti][e] = As[cur][mr][kk];
            }
        }
        #pragma unroll
        for (int tj = 0; tj < 4; ++tj) {
            int nc = waveN + tj * 16 + l15;
            #pragma unroll
            for (int e = 0; e < 16; ++e) {
                int kk = (e < 8 ? e : e + 8) + koff;
                bf[tj][e] = Bs[cur][nc][kk];
            }
        }
        #pragma unroll
        for (int ti = 0; ti < 2; ++ti)
            #pragma unroll
            for (int tj = 0; tj < 4; ++tj)
                acc[ti][tj] = __builtin_amdgcn_wmma_f32_16x16x32_f16(
                    false, af[ti], false, bf[tj], (short)0, acc[ti][tj],
                    false, false);
        __syncthreads();
    }

    // Epilogue per C/D layout: VGPR r -> rows (r, r+8) for lane halves
    int hi = lane >> 4;
    #pragma unroll
    for (int ti = 0; ti < 2; ++ti) {
        #pragma unroll
        for (int tj = 0; tj < 4; ++tj) {
            #pragma unroll
            for (int r = 0; r < 8; ++r) {
                int row = m0 + waveM + ti * 16 + r + hi * 8;
                int col = n0 + waveN + tj * 16 + l15;
                float v = acc[ti][tj][r];
                if (bias) v += bias[col];
                size_t o = (size_t)row * N + col;
                if (res) v += res[o];
                C[o] = v;
            }
        }
    }
}

// ---------------------------------------------------------------------------
// Host-side launch
// ---------------------------------------------------------------------------
static inline void gemm(hipStream_t st, const float* A, const float* B,
                        const float* bias, const float* res, float* C,
                        int M, int N, int K, int tanhA, int btrans) {
    dim3 g(N / TN, M / TM);
    k_gemm<<<g, 256, 0, st>>>(A, B, bias, res, C, M, N, K, tanhA, btrans);
}

extern "C" void kernel_launch(void* const* d_in, const int* in_sizes, int n_in,
                              void* d_out, int out_size, void* d_ws, size_t ws_size,
                              hipStream_t stream) {
    const float* tok_emb = (const float*)d_in[0];
    const float* pos_emb = (const float*)d_in[1];
    const float* emb_g   = (const float*)d_in[2];
    const float* emb_b   = (const float*)d_in[3];
    const float* Wq      = (const float*)d_in[4];
    const float* bq      = (const float*)d_in[5];
    const float* Wk      = (const float*)d_in[6];
    const float* bk      = (const float*)d_in[7];
    const float* Wv      = (const float*)d_in[8];
    const float* bv      = (const float*)d_in[9];
    const float* Wo      = (const float*)d_in[10];
    const float* bo      = (const float*)d_in[11];
    const float* J_attn  = (const float*)d_in[12];
    const float* lam_at  = (const float*)d_in[13];
    const float* an_g    = (const float*)d_in[14];
    const float* an_b    = (const float*)d_in[15];
    const float* W1      = (const float*)d_in[16];
    const float* b1      = (const float*)d_in[17];
    const float* J1      = (const float*)d_in[18];
    const float* lam1    = (const float*)d_in[19];
    const float* g1      = (const float*)d_in[20];
    const float* bb1     = (const float*)d_in[21];
    const float* W2      = (const float*)d_in[22];
    const float* b2      = (const float*)d_in[23];
    const float* J2      = (const float*)d_in[24];
    const float* lam2    = (const float*)d_in[25];
    const float* g2      = (const float*)d_in[26];
    const float* bb2     = (const float*)d_in[27];
    const float* fn_g    = (const float*)d_in[28];
    const float* fn_b    = (const float*)d_in[29];
    const float* fin_g   = (const float*)d_in[30];
    const float* fin_b   = (const float*)d_in[31];
    const float* lm_W    = (const float*)d_in[32];
    const int*   ids     = (const int*)d_in[33];

    const int NT = in_sizes[33];                 // B*S tokens (2048)

    // Workspace carve-up (fp32 activations)
    char* w = (char*)d_ws;
    auto carve = [&](size_t bytes) {
        char* p = w;
        w += (bytes + 255) & ~(size_t)255;
        return (float*)p;
    };
    float* X   = carve((size_t)NT * DD * 4);     // residual stream
    float* Aln = carve((size_t)NT * DD * 4);     // LN output
    float* Qb  = carve((size_t)NT * DD * 4);
    float* Kb  = carve((size_t)NT * DD * 4);
    float* Vb  = carve((size_t)NT * DD * 4);
    float* QJ  = carve((size_t)NT * DD * 4);
    float* Ob  = carve((size_t)NT * DD * 4);
    float* H1  = carve((size_t)NT * FF * 4);
    float* SJ1 = carve((size_t)NT * FF * 4);
    float* H2  = carve((size_t)NT * DD * 4);
    float* SJ2 = carve((size_t)NT * DD * 4);

    // Embedding + LN
    k_embed_ln<<<NT, 256, 0, stream>>>(tok_emb, pos_emb, emb_g, emb_b, ids, X);

    for (int l = 0; l < LL; ++l) {
        // ---- attention ----
        k_ln<<<NT, 256, 0, stream>>>(X, an_g + l * DD, an_b + l * DD, Aln, DD);
        gemm(stream, Aln, Wq + (size_t)l * DD * DD, bq + l * DD, nullptr, Qb, NT, DD, DD, 0, 0);
        gemm(stream, Aln, Wk + (size_t)l * DD * DD, bk + l * DD, nullptr, Kb, NT, DD, DD, 0, 0);
        gemm(stream, Aln, Wv + (size_t)l * DD * DD, bv + l * DD, nullptr, Vb, NT, DD, DD, 0, 0);
        k_qj<<<dim3(NT / 16, HH), 256, 0, stream>>>(Qb, J_attn + (size_t)l * HH * HD * HD, QJ);
        k_attn<<<dim3((NT / SS) * HH, SS / 32), 256, 0, stream>>>(
            Qb, Kb, Vb, QJ, lam_at, l, Ob);
        // x = x + o @ Wo + bo
        gemm(stream, Ob, Wo + (size_t)l * DD * DD, bo + l * DD, X, X, NT, DD, DD, 0, 0);

        // ---- QBNN FFN ----
        k_ln<<<NT, 256, 0, stream>>>(X, fn_g + l * DD, fn_b + l * DD, Aln, DD);
        // QBNN layer 1 (D -> FF)
        gemm(stream, Aln, W1 + (size_t)l * DD * FF, b1 + l * FF, nullptr, H1, NT, FF, DD, 0, 0);
        gemm(stream, Aln, J1 + (size_t)l * DD * FF, nullptr, nullptr, SJ1, NT, FF, DD, 1, 0);
        k_qbnn_post<<<NT, 256, 0, stream>>>(H1, SJ1, lam1, l, g1 + l * FF, bb1 + l * FF, H1, FF);
        // QBNN layer 2 (FF -> D)
        gemm(stream, H1, W2 + (size_t)l * FF * DD, b2 + l * DD, nullptr, H2, NT, DD, FF, 0, 0);
        gemm(stream, H1, J2 + (size_t)l * FF * DD, nullptr, nullptr, SJ2, NT, DD, FF, 1, 0);
        k_qbnn_post<<<NT, 256, 0, stream>>>(H2, SJ2, lam2, l, g2 + l * DD, bb2 + l * DD, H2, DD);
        // x = x + f
        k_add<<<(NT * DD + 255) / 256, 256, 0, stream>>>(X, H2, NT * DD);
    }

    // Final LN + LM head (x @ lm_W.T -> logits in d_out)
    k_ln<<<NT, 256, 0, stream>>>(X, fin_g, fin_b, Aln, DD);
    gemm(stream, Aln, lm_W, nullptr, nullptr, (float*)d_out, NT, VV, DD, 0, 1);
}
